// TtMultiHeadAttentionModel_72945724555279
// MI455X (gfx1250) — matrix operand order repaired and checked by reading, not compiled, not run on hardware
//
#include <hip/hip_runtime.h>

// ---------------------------------------------------------------------------
// MHA forward for B=8, S=1024, H=1024, NH=16, HD=64 on gfx1250 (wave32, WMMA)
// Path: fp32 -> bf16 convert -> fused QKV GEMM (v_wmma_f32_16x16x32_bf16,
//       branch-free software-pipelined fragment loads, peeled tail)
//       -> flash attention with online softmax (WMMA for QK^T and P*V).
// ---------------------------------------------------------------------------

typedef __bf16 v16bf __attribute__((ext_vector_type(16)));
typedef __bf16 v8bf  __attribute__((ext_vector_type(8)));
typedef __bf16 v4bf  __attribute__((ext_vector_type(4)));
typedef float  v8f   __attribute__((ext_vector_type(8)));

#define WMMA_BF16(A, Bm, Cm) \
  __builtin_amdgcn_wmma_f32_16x16x32_bf16(false, (A), false, (Bm), (short)0, (Cm), false, false)

static constexpr int kB  = 8;
static constexpr int kS  = 1024;
static constexpr int kH  = 1024;
static constexpr int kNH = 16;
static constexpr int kHD = 64;
static constexpr int kNROW = kB * kS;          // 8192
static constexpr int kLDSP = 40;               // padded LDS row stride (bf16), 80B = 5x16B

// Assemble a 16-wide A-fragment from two 16-byte chunks at p and p+8 elems.
__device__ inline v16bf load_a16(const __bf16* p) {
  v8bf lo = *(const v8bf*)(p);
  v8bf hi = *(const v8bf*)(p + 8);
  v16bf a;
#pragma unroll
  for (int e = 0; e < 8; ++e) { a[e] = lo[e]; a[8 + e] = hi[e]; }
  return a;
}

// ---------------------------------------------------------------------------
// fp32 -> bf16 conversion, vectorized by 4
// ---------------------------------------------------------------------------
__global__ __launch_bounds__(256) void cvt_f32_to_bf16(const float* __restrict__ in,
                                                       __bf16* __restrict__ out, int n4) {
  int i = blockIdx.x * blockDim.x + threadIdx.x;
  if (i >= n4) return;
  float4 v = ((const float4*)in)[i];
  v4bf o;
  o[0] = (__bf16)v.x; o[1] = (__bf16)v.y; o[2] = (__bf16)v.z; o[3] = (__bf16)v.w;
  ((v4bf*)out)[i] = o;
}

// ---------------------------------------------------------------------------
// Fused QKV projection: Y = X * W^T + b for W in {Wq, Wk, Wv}.
// X:[8192,1024] bf16 row-major, W:[1024,1024] bf16 row-major (both K-contig ->
// exactly the A x B(K-major) WMMA feeding pattern, no transpose needed).
// Block = 4 waves, 64x64 output tile per block; each wave: 16 rows x 64 cols.
// Steady-state loop is branch-free: fragments for the next (k,c) group are
// loaded unconditionally one group ahead; the final k-iteration is peeled so
// no guards/selects appear in the hot loop.
// Q,K stored [B*NH, S, HD] bf16; V stored transposed [B*NH, HD, S] bf16.
// ---------------------------------------------------------------------------
__global__ __launch_bounds__(128) void qkv_gemm(
    const __bf16* __restrict__ X,
    const __bf16* __restrict__ WQ, const __bf16* __restrict__ WK, const __bf16* __restrict__ WV,
    const float* __restrict__ BQ, const float* __restrict__ BK, const float* __restrict__ BV,
    __bf16* __restrict__ Qo, __bf16* __restrict__ Ko, __bf16* __restrict__ VTo) {
  const int lane   = threadIdx.x & 31;
  const int wave   = threadIdx.x >> 5;
  const int l15    = lane & 15;
  const int hiHalf = (lane & 16) ? 1 : 0;
  const int mBase  = blockIdx.x * 64 + wave * 16;   // row tile (16 rows / wave)
  const int h      = blockIdx.y;                    // head index == 64-col tile
  const int oBase  = h * 64;

  v8f accQ[4] = {}, accK[4] = {}, accV[4] = {};

  const int klo = hiHalf ? 16 : 0;   // A-frag K sub-offset
  const int kBo = hiHalf ? 16 : 0;   // B-frag K half offset

  // Hoisted per-lane fragment base pointers (k advances by uniform offset).
  const __bf16* ap = X + (size_t)(mBase + l15) * kH + klo;
  const __bf16* pq[4];
  const __bf16* pk[4];
  const __bf16* pv[4];
#pragma unroll
  for (int c = 0; c < 4; ++c) {
    const size_t wo = (size_t)(oBase + c * 16 + l15) * kH + kBo;
    pq[c] = WQ + wo; pk[c] = WK + wo; pv[c] = WV + wo;
  }

  // Prime the pipeline: A for k0=0 and B group (k0=0, c=0).
  v16bf a  = load_a16(ap);
  v16bf nq = *(const v16bf*)(pq[0]);
  v16bf nk = *(const v16bf*)(pk[0]);
  v16bf nv = *(const v16bf*)(pv[0]);

  // ---- main loop: k0 = 0 .. kH-64, all preloads unconditional ----
#pragma unroll 2
  for (int k0 = 0; k0 < kH - 32; k0 += 32) {
    const v16bf acur = a;
    a = load_a16(ap + k0 + 32);               // next A in flight early
#pragma unroll
    for (int c = 0; c < 4; ++c) {
      const v16bf bq = nq, bk = nk, bv = nv;
      // Issue next group's 6 b128 loads before consuming the current group.
      const int cn = (c + 1) & 3;
      const int kn = (c == 3) ? (k0 + 32) : k0;
      nq = *(const v16bf*)(pq[cn] + kn);
      nk = *(const v16bf*)(pk[cn] + kn);
      nv = *(const v16bf*)(pv[cn] + kn);
      accQ[c] = WMMA_BF16(acur, bq, accQ[c]);
      accK[c] = WMMA_BF16(acur, bk, accK[c]);
      accV[c] = WMMA_BF16(acur, bv, accV[c]);
    }
  }

  // ---- peeled final iteration (k0 = kH-32): compile-time-only guards ----
  {
    const int k0 = kH - 32;
    const v16bf acur = a;
#pragma unroll
    for (int c = 0; c < 4; ++c) {
      const v16bf bq = nq, bk = nk, bv = nv;
      if (c < 3) {
        nq = *(const v16bf*)(pq[c + 1] + k0);
        nk = *(const v16bf*)(pk[c + 1] + k0);
        nv = *(const v16bf*)(pv[c + 1] + k0);
      }
      accQ[c] = WMMA_BF16(acur, bq, accQ[c]);
      accK[c] = WMMA_BF16(acur, bk, accK[c]);
      accV[c] = WMMA_BF16(acur, bv, accV[c]);
    }
  }

  // Epilogue: bias add, bf16 store in head-split layouts (V transposed).
#pragma unroll
  for (int c = 0; c < 4; ++c) {
    const int d = c * 16 + l15;            // dim within head
    const int o = oBase + d;
    const float bq = BQ[o], bk = BK[o], bv = BV[o];
#pragma unroll
    for (int v = 0; v < 8; ++v) {
      const int row = mBase + v + hiHalf * 8;         // global (b*S + s)
      const int b   = row >> 10;
      const int s   = row & 1023;
      const size_t bh = (size_t)(b * kNH + h);
      Qo[(bh * kS + s) * kHD + d]  = (__bf16)(accQ[c][v] + bq);
      Ko[(bh * kS + s) * kHD + d]  = (__bf16)(accK[c][v] + bk);
      VTo[(bh * kHD + d) * kS + s] = (__bf16)(accV[c][v] + bv);
    }
  }
}

// ---------------------------------------------------------------------------
// Flash attention, one (b,head) per blockIdx.y, 128 query rows per block
// (8 waves x 16 rows). Online softmax; K chunks of 32 keys.
// scores: D = Q(16x64) x K^T via 2 WMMAs per 16-key tile.
// P re-layout (C-format -> A-format) through padded LDS (same-wave, in-order DS).
// ctx += P(16x32) x V(32x16) per 16-dim tile using transposed V (contig loads).
// ---------------------------------------------------------------------------
__global__ __launch_bounds__(256) void flash_attn(
    const __bf16* __restrict__ Qg, const __bf16* __restrict__ Kg,
    const __bf16* __restrict__ VTg, float* __restrict__ Out) {
  __shared__ __bf16 plds[8][16 * kLDSP];

  const int lane   = threadIdx.x & 31;
  const int wave   = threadIdx.x >> 5;
  const int l15    = lane & 15;
  const int hiHalf = (lane & 16) ? 1 : 0;
  const int bh     = blockIdx.y;                 // 0..127
  const int b      = bh >> 4;
  const int h      = bh & 15;
  const int qBase  = blockIdx.x * 128 + wave * 16;

  const __bf16* Qh = Qg  + (size_t)bh * kS * kHD;
  const __bf16* Kh = Kg  + (size_t)bh * kS * kHD;
  const __bf16* Vh = VTg + (size_t)bh * kHD * kS;

  const int klo = hiHalf ? 16 : 0;  // A-frag K sub-offset
  const int kBo = hiHalf ? 16 : 0;  // B-frag K half offset
  const int db  = hiHalf ? 16 : 0;  // score-B d half offset

  // Q fragments for this wave's 16 rows (K-dim = head dim 64 -> 2 frags), kept live.
  v16bf aq0, aq1;
  {
    const __bf16* qrow = Qh + (size_t)(qBase + l15) * kHD;
    aq0 = load_a16(qrow + klo);
    aq1 = load_a16(qrow + 32 + klo);
  }

  v8f acc[4] = {};
  float mst[8], lst[8];
#pragma unroll
  for (int v = 0; v < 8; ++v) { mst[v] = -__builtin_inff(); lst[v] = 0.0f; }

  const float scale = 0.03125f;  // 1/sqrt(H=1024)

  for (int kc = 0; kc < kS; kc += 32) {
    // Branchless prefetch of the next key/value chunk (wraps on last iter).
    const int kcn = (kc + 32) & (kS - 1);
    __builtin_prefetch(Kh + (size_t)(kcn + l15) * kHD, 0, 1);
    __builtin_prefetch(Vh + (size_t)l15 * kS + kcn, 0, 1);

    // ---- scores for keys [kc, kc+16) and [kc+16, kc+32) ----
    v8f s0 = {}, s1 = {};
    {
      const __bf16* k0p = Kh + (size_t)(kc + l15) * kHD + db;
      const __bf16* k1p = Kh + (size_t)(kc + 16 + l15) * kHD + db;
      v16bf b0 = *(const v16bf*)(k0p);        // d = db .. db+15
      v16bf b1 = *(const v16bf*)(k0p + 32);   // d = 32+db ..
      v16bf b2 = *(const v16bf*)(k1p);
      v16bf b3 = *(const v16bf*)(k1p + 32);
      s0 = WMMA_BF16(aq0, b0, s0);
      s0 = WMMA_BF16(aq1, b1, s0);
      s1 = WMMA_BF16(aq0, b2, s1);
      s1 = WMMA_BF16(aq1, b3, s1);
    }

    // ---- online softmax update (row reductions across 16-lane C groups) ----
    __bf16* pw = &plds[wave][0];
#pragma unroll
    for (int v = 0; v < 8; ++v) {
      float x0 = s0[v] * scale, x1 = s1[v] * scale;
      float t = fmaxf(x0, x1);
      t = fmaxf(t, __shfl_xor(t, 1, 32));
      t = fmaxf(t, __shfl_xor(t, 2, 32));
      t = fmaxf(t, __shfl_xor(t, 4, 32));
      t = fmaxf(t, __shfl_xor(t, 8, 32));
      const float mnew  = fmaxf(mst[v], t);
      const float alpha = __expf(mst[v] - mnew);
      mst[v] = mnew;
      const float p0 = __expf(x0 - mnew);
      const float p1 = __expf(x1 - mnew);
      float rs = p0 + p1;
      rs += __shfl_xor(rs, 1, 32);
      rs += __shfl_xor(rs, 2, 32);
      rs += __shfl_xor(rs, 4, 32);
      rs += __shfl_xor(rs, 8, 32);
      lst[v] = lst[v] * alpha + rs;
      // C-layout element (row q = v + 8*hiHalf, col = key) -> LDS [q][key]
      const int q = v + hiHalf * 8;
      pw[q * kLDSP + l15]      = (__bf16)p0;
      pw[q * kLDSP + 16 + l15] = (__bf16)p1;
#pragma unroll
      for (int c = 0; c < 4; ++c) acc[c][v] *= alpha;
    }

    // ---- read P back as 16x32 A-fragment (transposed lane mapping) ----
    // Same-wave LDS ops are in-order; padded stride keeps 16B alignment.
    const v16bf ap = load_a16(&plds[wave][l15 * kLDSP + klo]);

    // ---- ctx += P x V (V^T gives contiguous B-fragment loads) ----
#pragma unroll
    for (int c = 0; c < 4; ++c) {
      const int d = c * 16 + l15;
      v16bf bv = *(const v16bf*)(Vh + (size_t)d * kS + kc + kBo);
      acc[c] = WMMA_BF16(ap, bv, acc[c]);
    }
  }

  // ---- normalize and write fp32 output [B,S,H] ----
#pragma unroll
  for (int v = 0; v < 8; ++v) {
    const float inv  = 1.0f / lst[v];
    const int   srow = qBase + v + hiHalf * 8;
    float* orow = Out + ((size_t)(b * kS + srow)) * kH + h * kHD;
#pragma unroll
    for (int c = 0; c < 4; ++c) orow[c * 16 + l15] = acc[c][v] * inv;
  }
}

// ---------------------------------------------------------------------------
// Launcher
// ---------------------------------------------------------------------------
extern "C" void kernel_launch(void* const* d_in, const int* in_sizes, int n_in,
                              void* d_out, int out_size, void* d_ws, size_t ws_size,
                              hipStream_t stream) {
  const float* act = (const float*)d_in[0];
  const float* qw  = (const float*)d_in[1];
  const float* qb  = (const float*)d_in[2];
  const float* kw  = (const float*)d_in[3];
  const float* kb  = (const float*)d_in[4];
  const float* vw  = (const float*)d_in[5];
  const float* vb  = (const float*)d_in[6];
  float* out = (float*)d_out;

  // Workspace carve-up (bf16): act 16MB, 3 weights 6MB, Q/K/V^T 48MB -> ~70MB.
  char* ws = (char*)d_ws;
  size_t off = 0;
  __bf16* act_bf = (__bf16*)(ws + off); off += (size_t)kNROW * kH * 2;
  __bf16* qw_bf  = (__bf16*)(ws + off); off += (size_t)kH * kH * 2;
  __bf16* kw_bf  = (__bf16*)(ws + off); off += (size_t)kH * kH * 2;
  __bf16* vw_bf  = (__bf16*)(ws + off); off += (size_t)kH * kH * 2;
  __bf16* q_ws   = (__bf16*)(ws + off); off += (size_t)kNROW * kH * 2;
  __bf16* k_ws   = (__bf16*)(ws + off); off += (size_t)kNROW * kH * 2;
  __bf16* vt_ws  = (__bf16*)(ws + off); off += (size_t)kNROW * kH * 2;
  (void)ws_size; (void)in_sizes; (void)n_in; (void)out_size;

  // 1) fp32 -> bf16 conversions
  {
    int n4 = (kNROW * kH) / 4;  // 2,097,152
    cvt_f32_to_bf16<<<(n4 + 255) / 256, 256, 0, stream>>>(act, act_bf, n4);
    int w4 = (kH * kH) / 4;     // 262,144
    cvt_f32_to_bf16<<<(w4 + 255) / 256, 256, 0, stream>>>(qw, qw_bf, w4);
    cvt_f32_to_bf16<<<(w4 + 255) / 256, 256, 0, stream>>>(kw, kw_bf, w4);
    cvt_f32_to_bf16<<<(w4 + 255) / 256, 256, 0, stream>>>(vw, vw_bf, w4);
  }

  // 2) fused QKV projection: grid = 128 row-tiles x 16 heads, 4 waves/block
  {
    dim3 grid(kNROW / 64, kNH);
    qkv_gemm<<<grid, 128, 0, stream>>>(act_bf, qw_bf, kw_bf, vw_bf,
                                       qb, kb, vb, q_ws, k_ws, vt_ws);
  }

  // 3) flash attention: grid = 8 query-tiles x 128 (b,head), 8 waves/block
  {
    dim3 grid(kS / 128, kB * kNH);
    flash_attn<<<grid, 256, 0, stream>>>(q_ws, k_ws, vt_ws, out);
  }
}
